// SimpleMPGNN_41875931136489
// MI455X (gfx1250) — matrix operand order repaired and checked by reading, not compiled, not run on hardware
//
#include <hip/hip_runtime.h>
#include <hip/hip_bf16.h>

// ---------------------------------------------------------------------------
// SimpleMPGNN on MI455X (gfx1250), wave32 + bf16 WMMA.
// Key algebraic reductions:
//   * edge MLP depends only on (dst,src) pair -> evaluate 128x128 pairs, not 524288 edges
//   * first EdgeConv linear decomposes: concat(xi, xj-xi)W1 = xi(Wt-Wb) + xj Wb
// Heavy GEMMs: [16384,512]x[512,512] per layer in v_wmma_f32_16x16x32_bf16.
// Star kernel: A-slice staged in LDS; ks kept as a rolled loop (unroll 1) so
// LICM cannot hoist A-tiles across the ci loop (previous rounds spilled);
// B-tile double-buffered from global; 8 independent accumulators.
// ---------------------------------------------------------------------------

typedef __attribute__((ext_vector_type(16))) __bf16 v16bf;
typedef __attribute__((ext_vector_type(8)))  float  v8f;

#define HF 512

// workspace byte offsets (256B aligned)
#define WS_MASK   0                         // 128 * 4 u32 = 2048 B
#define WS_G0     2048                      // 512 f32
#define WS_A      4096                      // 128x512 f32
#define WS_B      (WS_A + 262144)           // 128x512 f32
#define WS_H1     (WS_B + 262144)           // 128x512 f32
#define WS_H2     (WS_H1 + 262144)          // 128x512 f32
#define WS_BPACK  (WS_H2 + 262144)          // 512x512 bf16 packed
#define WS_UPACK  (WS_BPACK + 524288)       // 16384x512 bf16 packed (16 MB)

__device__ __forceinline__ unsigned short f2bf(float f) {
  unsigned u = __builtin_bit_cast(unsigned, f);
  u += 0x7FFFu + ((u >> 16) & 1u);          // round-to-nearest-even
  return (unsigned short)(u >> 16);
}

// ---- zero mask + g0 accumulator ------------------------------------------
__global__ void k_zero(unsigned* __restrict__ mask, float* __restrict__ g0) {
  int i = blockIdx.x * blockDim.x + threadIdx.x;
  if (i < 512) mask[i] = 0u;
  else if (i < 1024) g0[i - 512] = 0.f;
}

// ---- build (dst,src) presence bitmask ------------------------------------
__global__ void k_mask(const int* __restrict__ ei, unsigned* __restrict__ mask, int E) {
  int e = blockIdx.x * blockDim.x + threadIdx.x;
  if (e >= E) return;
  int s = ei[e];         // src
  int d = ei[E + e];     // dst
  atomicOr(&mask[d * 4 + (s >> 5)], 1u << (s & 31));
}

// ---- node projections: A = x(Wtop-Wbot), B = x Wbot ----------------------
__global__ void k_nodeproj(const float* __restrict__ x, const float* __restrict__ Wtop,
                           const float* __restrict__ Wbot, float* __restrict__ A,
                           float* __restrict__ B, int K) {
  int d = blockIdx.x, n = threadIdx.x;      // 128 blocks x 512 threads
  float accA = 0.f, accB = 0.f;
  for (int k = 0; k < K; ++k) {
    float xv = x[d * K + k];
    float wt = Wtop[k * HF + n];
    float wb = Wbot[k * HF + n];
    accA += xv * (wt - wb);
    accB += xv * wb;
  }
  A[d * HF + n] = accA;
  B[d * HF + n] = accB;
}

// ---- pair activations, written directly in WMMA 16-bit A-matrix layout ---
// Upack[d][st][ks][lane][e], e=0..15 bf16 (32B contiguous per lane per kstep)
__global__ void k_upack(const float* __restrict__ A, const float* __restrict__ B,
                        const float* __restrict__ b1, unsigned short* __restrict__ U) {
  int idx = blockIdx.x * blockDim.x + threadIdx.x;   // 8,388,608 elements
  int e    = idx & 15;
  int lane = (idx >> 4) & 31;
  int ks   = (idx >> 9) & 15;
  int st   = (idx >> 13) & 7;
  int d    = idx >> 16;
  int hi   = lane >> 4;
  int s    = st * 16 + (lane & 15);                  // matrix row (src node)
  // ISA 16-bit A layout: elems 0..7 -> K = hi*8 + e ; elems 8..15 -> K = 16 + hi*8 + (e-8)
  int k = ks * 32 + ((e < 8) ? (hi * 8 + e) : (16 + hi * 8 + (e - 8)));
  float v = A[d * HF + k] + B[s * HF + k] + b1[k];
  U[idx] = f2bf(fmaxf(v, 0.f));
}

// ---- W2 repack into WMMA 16-bit B-matrix (32x16) layout ------------------
// Bpack[ks][ct][lane][e]: n = ct*16 + (lane&15), k = ks*32 + (e<8 ? (lane>>4)*8 + e : 16 + (lane>>4)*8 + (e-8))
__global__ void k_bpack(const float* __restrict__ W2, unsigned short* __restrict__ Bp) {
  int idx = blockIdx.x * blockDim.x + threadIdx.x;   // 262,144 elements
  int e    = idx & 15;
  int lane = (idx >> 4) & 31;
  int ct   = (idx >> 9) & 31;
  int ks   = idx >> 14;
  int n = ct * 16 + (lane & 15);
  int k = ks * 32 + ((e < 8) ? ((lane >> 4) * 8 + e) : (16 + (lane >> 4) * 8 + (e - 8)));
  Bp[idx] = f2bf(W2[k * HF + n]);
}

// ---- star kernel: GEMM2 + masked segment-max + bias + relu ---------------
// one block per dst node d; 256 threads = 8 waves; wave handles 4 col-tiles.
__global__ void k_edge_gemm_max(const unsigned short* __restrict__ Upack,
                                const unsigned short* __restrict__ Bpack,
                                const float* __restrict__ b2,
                                const unsigned* __restrict__ mask,
                                float* __restrict__ hout) {
  extern __shared__ unsigned short sU[];   // 8*16*32*16 = 65536 ushort = 128 KB
  const int d    = blockIdx.x;
  const int tid  = threadIdx.x;
  const int lane = tid & 31;
  const int wave = tid >> 5;
  const int hi   = lane >> 4;
  const int nc   = lane & 15;

  // cooperative staging of this d's A-slice (131072 B) into LDS
  {
    const uint4* gsrc = reinterpret_cast<const uint4*>(Upack + (size_t)d * 65536);
    uint4* ldst = reinterpret_cast<uint4*>(sU);
    #pragma unroll
    for (int i = 0; i < 32; ++i) ldst[tid + i * 256] = gsrc[tid + i * 256];
  }
  __syncthreads();

  const unsigned mw0 = mask[d * 4 + 0];
  const unsigned mw1 = mask[d * 4 + 1];
  const unsigned mw2 = mask[d * 4 + 2];
  const unsigned mw3 = mask[d * 4 + 3];
  const float NEG_INF = -__builtin_inff();
  const v8f vzero = {};

  #pragma unroll 1
  for (int ci = 0; ci < 4; ++ci) {
    const int ct = wave + ci * 8;                    // col tile 0..31
    v8f acc[8];
    #pragma unroll
    for (int st = 0; st < 8; ++st) acc[st] = vzero;

    const unsigned short* Bb = Bpack + ((size_t)ct * 32 + lane) * 16;
    // double-buffered B-tile; ks stays a rolled loop so A-tiles cannot be
    // hoisted/spilled across the ci loop.
    v16bf b = *reinterpret_cast<const v16bf*>(Bb);
    #pragma unroll 1
    for (int ks = 0; ks < 16; ++ks) {                // K = 512
      const int ksn = (ks + 1 < 16) ? (ks + 1) : 15;
      v16bf bn = *reinterpret_cast<const v16bf*>(Bb + (size_t)ksn * (32 * 32 * 16));
      #pragma unroll
      for (int st = 0; st < 8; ++st) {
        v16bf a = *reinterpret_cast<const v16bf*>(sU + (((st * 16 + ks) * 32 + lane) * 16));
        acc[st] = __builtin_amdgcn_wmma_f32_16x16x32_bf16(
            false, a, false, b, (short)0, acc[st], false, false);
      }
      b = bn;
    }

    // masked segment-max over all 128 src rows
    float m = NEG_INF;
    #pragma unroll
    for (int st = 0; st < 8; ++st) {
      // s-tile st spans s = st*16 .. st*16+15 -> single 32-bit mask word st>>1
      const int q = st >> 1;
      const unsigned mword = (q == 0) ? mw0 : ((q == 1) ? mw1 : ((q == 2) ? mw2 : mw3));
      #pragma unroll
      for (int i = 0; i < 8; ++i) {                  // C/D layout: row = i + hi*8
        int s = st * 16 + i + hi * 8;
        bool present = (mword >> (s & 31)) & 1u;
        float v = present ? acc[st][i] : NEG_INF;
        m = fmaxf(m, v);
      }
    }
    m = fmaxf(m, __shfl_xor(m, 16, 32));             // combine lane halves (same col)
    // empty segment: m=-inf -> +b2 stays -inf -> max(.,0)=0 (matches ref fill+relu)
    float h = fmaxf(m + b2[ct * 16 + nc], 0.f);
    if (lane < 16) hout[d * HF + ct * 16 + lane] = h;
  }
}

// ---- big GEMV: g0 += v[65536] . lin_W[65536,512] (memory-bound, 134 MB) --
__global__ void k_gemv_big(const float* __restrict__ v, const float* __restrict__ W,
                           float* __restrict__ g0) {
  int n = threadIdx.x;                               // 512
  int b = blockIdx.x;                                // 128 m-chunks of 512
  int m0 = b * 512;
  float acc = 0.f;
  for (int mm = 0; mm < 512; ++mm)
    acc += v[m0 + mm] * W[(size_t)(m0 + mm) * 512 + n];
  atomicAdd(&g0[n], acc);
}

// ---- fused head tail: bias+relu -> 512->256 -> 256->128 -> softmax -------
__global__ void k_head(const float* __restrict__ g0, const float* __restrict__ lin_b,
                       const float* __restrict__ lin1_W, const float* __restrict__ lin1_b,
                       const float* __restrict__ out_W, const float* __restrict__ out_b,
                       float* __restrict__ out) {
  __shared__ float sg0[512];
  __shared__ float sg1[256];
  __shared__ float sl[128];
  __shared__ float red[128];
  int t = threadIdx.x;                               // 256 threads
  sg0[t]       = fmaxf(g0[t]       + lin_b[t],       0.f);
  sg0[t + 256] = fmaxf(g0[t + 256] + lin_b[t + 256], 0.f);
  __syncthreads();
  float a1 = lin1_b[t];
  for (int k = 0; k < 512; ++k) a1 += sg0[k] * lin1_W[k * 256 + t];
  sg1[t] = fmaxf(a1, 0.f);
  __syncthreads();
  if (t < 128) {
    float a2 = out_b[t];
    for (int k = 0; k < 256; ++k) a2 += sg1[k] * out_W[k * 128 + t];
    sl[t] = fmaxf(a2, 0.f);
  }
  __syncthreads();
  if (t < 128) red[t] = sl[t];
  __syncthreads();
  for (int s = 64; s > 0; s >>= 1) { if (t < s) red[t] = fmaxf(red[t], red[t + s]); __syncthreads(); }
  float mx = red[0];
  __syncthreads();
  if (t < 128) { float e = __expf(sl[t] - mx); sl[t] = e; red[t] = e; }
  __syncthreads();
  for (int s = 64; s > 0; s >>= 1) { if (t < s) red[t] += red[t + s]; __syncthreads(); }
  if (t < 128) out[t] = sl[t] / red[0];
}

// ---------------------------------------------------------------------------
extern "C" void kernel_launch(void* const* d_in, const int* in_sizes, int n_in,
                              void* d_out, int out_size, void* d_ws, size_t ws_size,
                              hipStream_t stream) {
  const float* x     = (const float*)d_in[0];
  const int*   ei    = (const int*)d_in[1];
  const float* c1W1  = (const float*)d_in[2];
  const float* c1b1  = (const float*)d_in[3];
  const float* c1W2  = (const float*)d_in[4];
  const float* c1b2  = (const float*)d_in[5];
  const float* c2W1  = (const float*)d_in[6];
  const float* c2b1  = (const float*)d_in[7];
  const float* c2W2  = (const float*)d_in[8];
  const float* c2b2  = (const float*)d_in[9];
  const float* linW  = (const float*)d_in[10];
  const float* linb  = (const float*)d_in[11];
  const float* lin1W = (const float*)d_in[12];
  const float* lin1b = (const float*)d_in[13];
  const float* outW  = (const float*)d_in[14];
  const float* outb  = (const float*)d_in[15];

  char* ws = (char*)d_ws;
  unsigned*       mask  = (unsigned*)(ws + WS_MASK);
  float*          g0    = (float*)(ws + WS_G0);
  float*          Abuf  = (float*)(ws + WS_A);
  float*          Bbuf  = (float*)(ws + WS_B);
  float*          h1    = (float*)(ws + WS_H1);
  float*          h2    = (float*)(ws + WS_H2);
  unsigned short* Bpack = (unsigned short*)(ws + WS_BPACK);
  unsigned short* Upack = (unsigned short*)(ws + WS_UPACK);

  const int E = in_sizes[1] / 2;
  const int LDS_BYTES = 131072;   // 128 KB A-slice per block

  k_zero<<<4, 256, 0, stream>>>(mask, g0);
  k_mask<<<(E + 255) / 256, 256, 0, stream>>>(ei, mask, E);

  // EdgeConv layer 1 (F_IN = 256, so 2C = 512 rows in W1: top 256 / bottom 256)
  k_nodeproj<<<128, 512, 0, stream>>>(x, c1W1, c1W1 + 256 * 512, Abuf, Bbuf, 256);
  k_upack<<<32768, 256, 0, stream>>>(Abuf, Bbuf, c1b1, Upack);
  k_bpack<<<1024, 256, 0, stream>>>(c1W2, Bpack);
  k_edge_gemm_max<<<128, 256, LDS_BYTES, stream>>>(Upack, Bpack, c1b2, mask, h1);

  // EdgeConv layer 2 (2C = 1024 rows in W1: top 512 / bottom 512)
  k_nodeproj<<<128, 512, 0, stream>>>(h1, c2W1, c2W1 + 512 * 512, Abuf, Bbuf, 512);
  k_upack<<<32768, 256, 0, stream>>>(Abuf, Bbuf, c2b1, Upack);
  k_bpack<<<1024, 256, 0, stream>>>(c2W2, Bpack);
  k_edge_gemm_max<<<128, 256, LDS_BYTES, stream>>>(Upack, Bpack, c2b2, mask, h2);

  // dense head: v = flatten(h2) row-major == reshape(-1)
  k_gemv_big<<<128, 512, 0, stream>>>(h2, linW, g0);
  k_head<<<1, 256, 0, stream>>>(g0, linb, lin1W, lin1b, outW, outb, (float*)d_out);
}
